// ffMoE_26276609917410
// MI455X (gfx1250) — matrix operand order repaired and checked by reading, not compiled, not run on hardware
//
#include <hip/hip_runtime.h>
#include <hip/hip_bf16.h>

// ---------------- problem constants ----------------
static constexpr int E = 8;
static constexpr int TOPK = 2;
static constexpr int H = 768;
static constexpr int F = 3072;
static constexpr int T = 4096;          // B*S tokens
static constexpr int A_CAP = 8448;      // T*TOPK padded per-expert to 32 (8192 + 8*31 <= 8448)
static constexpr int MAX_TILES = A_CAP / 32;  // 264

// ---------------- bf16 / WMMA fragment types ----------------
typedef __bf16 bf16_t;
typedef __bf16 v16bf __attribute__((ext_vector_type(16)));
typedef __bf16 bf16x8 __attribute__((ext_vector_type(8)));
typedef float  v8f   __attribute__((ext_vector_type(8)));

union FragBF { v16bf v; bf16x8 h[2]; };

struct alignas(8) bhalf4 { bf16_t x, y, z, w; };

// ---------------- workspace layout (bytes) ----------------
static constexpr size_t SZ_WT     = (size_t)E * F * H * 2;          // 37,748,736
static constexpr size_t OFF_W1T   = 0;
static constexpr size_t OFF_W2T   = OFF_W1T + SZ_WT;
static constexpr size_t OFF_XG    = OFF_W2T + SZ_WT;                // bf16 [A_CAP][H]
static constexpr size_t OFF_H1    = OFF_XG + (size_t)A_CAP * H * 2; // bf16 [A_CAP][F]
static constexpr size_t OFF_GATES = OFF_H1 + (size_t)A_CAP * F * 2; // f32 [A_CAP]
static constexpr size_t OFF_TOK   = OFF_GATES + (size_t)A_CAP * 4;  // i32 [A_CAP]
static constexpr size_t OFF_TKE   = OFF_TOK + (size_t)A_CAP * 4;    // i32 [T][2]
static constexpr size_t OFF_TKG   = OFF_TKE + (size_t)T * TOPK * 4; // f32 [T][2]
static constexpr size_t OFF_CNT   = OFF_TKG + (size_t)T * TOPK * 4; // i32 [E]
static constexpr size_t OFF_OFFS  = OFF_CNT + 64;                   // i32 [E+1]
static constexpr size_t OFF_CURS  = OFF_OFFS + 64;                  // i32 [E]
static constexpr size_t OFF_PSUM  = OFF_CURS + 64;                  // f32 [E]
static constexpr size_t OFF_T1C   = OFF_PSUM + 64;                  // f32 [E]
static constexpr size_t OFF_SCHN  = OFF_T1C + 64;                   // i32
static constexpr size_t OFF_SCHED = OFF_SCHN + 64;                  // int4 [MAX_TILES]

// ---------------- kernel: zero init ----------------
__global__ void moe_zinit(unsigned int* xg32, int nxg32, int* tok, float* gates,
                          int* counts, float* psum, float* t1c, float* out, int nout) {
    int idx = blockIdx.x * blockDim.x + threadIdx.x;
    int stride = gridDim.x * blockDim.x;
    for (int i = idx; i < nxg32; i += stride) xg32[i] = 0u;
    for (int i = idx; i < A_CAP; i += stride) { tok[i] = -1; gates[i] = 0.0f; }
    for (int i = idx; i < nout; i += stride) out[i] = 0.0f;
    if (idx < E) { counts[idx] = 0; psum[idx] = 0.0f; t1c[idx] = 0.0f; }
}

// ---------------- kernel: fp32 -> bf16 transpose (per expert slice) ----------
// src: [R][C] fp32 row-major; dst: [C][R] bf16 row-major
__global__ __launch_bounds__(256) void moe_transpose(const float* __restrict__ src,
                                                     bf16_t* __restrict__ dst,
                                                     int R, int C) {
    __shared__ float tile[32][33];
    size_t eoff = (size_t)blockIdx.z * R * C;
    src += eoff; dst += eoff;
    int x0 = blockIdx.x * 32, y0 = blockIdx.y * 32;
    int tx = threadIdx.x, ty = threadIdx.y;
#pragma unroll
    for (int i = 0; i < 4; ++i)
        tile[ty + 8 * i][tx] = src[(size_t)(y0 + ty + 8 * i) * C + (x0 + tx)];
    __syncthreads();
#pragma unroll
    for (int i = 0; i < 4; ++i)
        dst[(size_t)(x0 + ty + 8 * i) * R + (y0 + tx)] = (bf16_t)tile[tx][ty + 8 * i];
}

// ---------------- kernel: router (1 wave per token) ----------------
__global__ __launch_bounds__(256) void moe_router(const float* __restrict__ x,
                                                  const float* __restrict__ rw,
                                                  const float* __restrict__ rb,
                                                  int* __restrict__ tke, float* __restrict__ tkg,
                                                  int* __restrict__ counts,
                                                  float* __restrict__ psum,
                                                  float* __restrict__ t1c) {
    int wave = threadIdx.x >> 5, lane = threadIdx.x & 31;
    int t = blockIdx.x * 8 + wave;
    if (t >= T) return;
    float s[E];
#pragma unroll
    for (int e = 0; e < E; ++e) s[e] = 0.0f;
    const float* xr = x + (size_t)t * H;
    for (int h = lane; h < H; h += 32) {
        float xv = xr[h];
        const float4* w4 = (const float4*)(rw + (size_t)h * E);
        float4 w0 = w4[0], w1 = w4[1];
        s[0] += xv * w0.x; s[1] += xv * w0.y; s[2] += xv * w0.z; s[3] += xv * w0.w;
        s[4] += xv * w1.x; s[5] += xv * w1.y; s[6] += xv * w1.z; s[7] += xv * w1.w;
    }
#pragma unroll
    for (int m = 16; m >= 1; m >>= 1)
#pragma unroll
        for (int e = 0; e < E; ++e) s[e] += __shfl_xor(s[e], m, 32);
    if (lane == 0) {
        float p[E];
        float mx = -1e30f;
#pragma unroll
        for (int e = 0; e < E; ++e) { p[e] = s[e] + rb[e]; mx = p[e] > mx ? p[e] : mx; }
        float den = 0.0f;
#pragma unroll
        for (int e = 0; e < E; ++e) { p[e] = __expf(p[e] - mx); den += p[e]; }
        float inv = 1.0f / den;
#pragma unroll
        for (int e = 0; e < E; ++e) p[e] *= inv;
        int i0 = 0;
#pragma unroll
        for (int e = 1; e < E; ++e) if (p[e] > p[i0]) i0 = e;
        int i1 = (i0 == 0) ? 1 : 0;
#pragma unroll
        for (int e = 0; e < E; ++e) if (e != i0 && p[e] > p[i1]) i1 = e;
        float g0 = p[i0], g1 = p[i1];
        float dsum = 1.0f / (g0 + g1 + 1e-9f);
        tke[t * 2 + 0] = i0; tke[t * 2 + 1] = i1;
        tkg[t * 2 + 0] = g0 * dsum; tkg[t * 2 + 1] = g1 * dsum;
        atomicAdd(&counts[i0], 1);
        atomicAdd(&counts[i1], 1);
        unsafeAtomicAdd(&t1c[i0], 1.0f);
#pragma unroll
        for (int e = 0; e < E; ++e) unsafeAtomicAdd(&psum[e], p[e]);
    }
}

// ---------------- kernel: scan + tile schedule + balance loss -------------
__global__ void moe_scan(const int* __restrict__ counts, int* __restrict__ offs,
                         int* __restrict__ curs, int4* __restrict__ sched,
                         int* __restrict__ schn, const float* __restrict__ psum,
                         const float* __restrict__ t1c, float* __restrict__ loss_out) {
    if (threadIdx.x == 0 && blockIdx.x == 0) {
        int off = 0, idx = 0;
        for (int e = 0; e < E; ++e) {
            offs[e] = off; curs[e] = 0;
            int c = counts[e];
            for (int m = 0; m < c; m += 32)
                sched[idx++] = make_int4(e, off + m, off + c, 0);
            off += (c + 31) & ~31;
        }
        offs[E] = off;
        *schn = idx;
        float bal = 0.0f;
        const float invT = 1.0f / (float)T;
        for (int e = 0; e < E; ++e) bal += (t1c[e] * invT) * (psum[e] * invT);
        *loss_out = 0.001f * bal;
    }
}

// ---------------- kernel: scatter tokens (1 wave per assignment) ----------
__global__ __launch_bounds__(256) void moe_scatter(const float* __restrict__ x,
                                                   const int* __restrict__ tke,
                                                   const float* __restrict__ tkg,
                                                   const int* __restrict__ offs,
                                                   int* __restrict__ curs,
                                                   bf16_t* __restrict__ xg,
                                                   int* __restrict__ tok,
                                                   float* __restrict__ gates) {
    int wave = blockIdx.x * 8 + (threadIdx.x >> 5);
    int lane = threadIdx.x & 31;
    int t = wave >> 1, k = wave & 1;
    if (t >= T) return;
    int e = tke[t * 2 + k];
    float g = tkg[t * 2 + k];
    int pos = 0;
    if (lane == 0) pos = offs[e] + atomicAdd(&curs[e], 1);
    pos = __shfl(pos, 0, 32);
    const float4* src = (const float4*)(x + (size_t)t * H);
    bhalf4* dst = (bhalf4*)(xg + (size_t)pos * H);
    for (int i = lane; i < H / 4; i += 32) {
        float4 v = src[i];
        bhalf4 o = { (bf16_t)v.x, (bf16_t)v.y, (bf16_t)v.z, (bf16_t)v.w };
        dst[i] = o;
    }
    if (lane == 0) { tok[pos] = t; gates[pos] = g; }
}

// ---------------- kernel: stage-1 grouped GEMM, h1 = relu^2(xg*w1 + b1) ---
// block = 256 thr (8 waves); tile 32(M) x 64(N); wave -> one 16x16 subtile
__global__ __launch_bounds__(256) void moe_ffn1(const bf16_t* __restrict__ xg,
                                                const bf16_t* __restrict__ w1t,
                                                const float* __restrict__ b1,
                                                bf16_t* __restrict__ h1,
                                                const int4* __restrict__ sched,
                                                const int* __restrict__ schn) {
    if ((int)blockIdx.x >= *schn) return;
    int4 s = sched[blockIdx.x];
    int e = s.x, row0 = s.y;
    int n0 = blockIdx.y * 64;
    int lane = threadIdx.x & 31, wave = threadIdx.x >> 5;
    int m_off = (wave & 1) * 16, n_off = (wave >> 1) * 16;
    int r = lane & 15, half = lane >> 4;

    const bf16_t* aRow = xg + (size_t)(row0 + m_off + r) * H + half * 8;
    const bf16_t* bRow = w1t + ((size_t)e * F + n0 + n_off + r) * H + half * 8;

    v8f acc = {};
#pragma unroll 4
    for (int k = 0; k < H; k += 32) {
        FragBF a, b;
        a.h[0] = *(const bf16x8*)(aRow + k);
        a.h[1] = *(const bf16x8*)(aRow + k + 16);
        b.h[0] = *(const bf16x8*)(bRow + k);
        b.h[1] = *(const bf16x8*)(bRow + k + 16);
        acc = __builtin_amdgcn_wmma_f32_16x16x32_bf16(false, a.v, false, b.v,
                                                      (short)0, acc, false, false);
    }
    int col = n0 + n_off + r;
    float bias = b1[e * F + col];
    int rbase = row0 + m_off + half * 8;
#pragma unroll
    for (int i = 0; i < 8; ++i) {
        float v = acc[i] + bias;
        v = v > 0.0f ? v : 0.0f;
        v = v * v;
        h1[(size_t)(rbase + i) * F + col] = (bf16_t)v;
    }
}

// ---------------- kernel: stage-2 grouped GEMM, out += gate*(h1*w2 + b2) --
__global__ __launch_bounds__(256) void moe_ffn2(const bf16_t* __restrict__ h1,
                                                const bf16_t* __restrict__ w2t,
                                                const float* __restrict__ b2,
                                                const int* __restrict__ tok,
                                                const float* __restrict__ gates,
                                                float* __restrict__ out,
                                                const int4* __restrict__ sched,
                                                const int* __restrict__ schn) {
    if ((int)blockIdx.x >= *schn) return;
    int4 s = sched[blockIdx.x];
    int e = s.x, row0 = s.y;
    int n0 = blockIdx.y * 64;
    int lane = threadIdx.x & 31, wave = threadIdx.x >> 5;
    int m_off = (wave & 1) * 16, n_off = (wave >> 1) * 16;
    int r = lane & 15, half = lane >> 4;

    const bf16_t* aRow = h1 + (size_t)(row0 + m_off + r) * F + half * 8;
    const bf16_t* bRow = w2t + ((size_t)e * H + n0 + n_off + r) * F + half * 8;

    v8f acc = {};
#pragma unroll 4
    for (int k = 0; k < F; k += 32) {
        FragBF a, b;
        a.h[0] = *(const bf16x8*)(aRow + k);
        a.h[1] = *(const bf16x8*)(aRow + k + 16);
        b.h[0] = *(const bf16x8*)(bRow + k);
        b.h[1] = *(const bf16x8*)(bRow + k + 16);
        acc = __builtin_amdgcn_wmma_f32_16x16x32_bf16(false, a.v, false, b.v,
                                                      (short)0, acc, false, false);
    }
    int col = n0 + n_off + r;
    float bias = b2[e * H + col];
    int rbase = row0 + m_off + half * 8;
#pragma unroll
    for (int i = 0; i < 8; ++i) {
        int a = rbase + i;
        int t = tok[a];
        if (t >= 0) {
            float val = gates[a] * (acc[i] + bias);
            unsafeAtomicAdd(&out[(size_t)t * H + col], val);
        }
    }
}

// ---------------- host launcher ----------------
extern "C" void kernel_launch(void* const* d_in, const int* in_sizes, int n_in,
                              void* d_out, int out_size, void* d_ws, size_t ws_size,
                              hipStream_t stream) {
    const float* x  = (const float*)d_in[0];
    const float* rw = (const float*)d_in[1];
    const float* rb = (const float*)d_in[2];
    const float* w1 = (const float*)d_in[3];
    const float* b1 = (const float*)d_in[4];
    const float* w2 = (const float*)d_in[5];
    const float* b2 = (const float*)d_in[6];
    float* out = (float*)d_out;                 // [T*H] out + [1] bal_loss

    char* ws = (char*)d_ws;
    bf16_t* w1t   = (bf16_t*)(ws + OFF_W1T);
    bf16_t* w2t   = (bf16_t*)(ws + OFF_W2T);
    bf16_t* xg    = (bf16_t*)(ws + OFF_XG);
    bf16_t* h1    = (bf16_t*)(ws + OFF_H1);
    float*  gates = (float*)(ws + OFF_GATES);
    int*    tok   = (int*)(ws + OFF_TOK);
    int*    tke   = (int*)(ws + OFF_TKE);
    float*  tkg   = (float*)(ws + OFF_TKG);
    int*    cnt   = (int*)(ws + OFF_CNT);
    int*    offs  = (int*)(ws + OFF_OFFS);
    int*    curs  = (int*)(ws + OFF_CURS);
    float*  psum  = (float*)(ws + OFF_PSUM);
    float*  t1c   = (float*)(ws + OFF_T1C);
    int*    schn  = (int*)(ws + OFF_SCHN);
    int4*   sched = (int4*)(ws + OFF_SCHED);

    const int nout = T * H + 1;
    const int nxg32 = A_CAP * H / 2;

    moe_zinit<<<2048, 256, 0, stream>>>((unsigned int*)xg, nxg32, tok, gates,
                                        cnt, psum, t1c, out, nout);
    // w1 [E][H][F] -> w1t [E][F][H] bf16 ; w2 [E][F][H] -> w2t [E][H][F] bf16
    moe_transpose<<<dim3(F / 32, H / 32, E), dim3(32, 8), 0, stream>>>(w1, w1t, H, F);
    moe_transpose<<<dim3(H / 32, F / 32, E), dim3(32, 8), 0, stream>>>(w2, w2t, F, H);

    moe_router<<<T / 8, 256, 0, stream>>>(x, rw, rb, tke, tkg, cnt, psum, t1c);
    moe_scan<<<1, 32, 0, stream>>>(cnt, offs, curs, sched, schn, psum, t1c, out + (size_t)T * H);
    moe_scatter<<<T * TOPK / 8, 256, 0, stream>>>(x, tke, tkg, offs, curs, xg, tok, gates);

    moe_ffn1<<<dim3(MAX_TILES, F / 64), 256, 0, stream>>>(xg, w1t, b1, h1, sched, schn);
    moe_ffn2<<<dim3(MAX_TILES, H / 64), 256, 0, stream>>>(h1, w2t, b2, tok, gates, out, sched, schn);
}
